// CausalSelfAttention_75883482185738
// MI455X (gfx1250) — matrix operand order repaired and checked by reading, not compiled, not run on hardware
//
#include <hip/hip_runtime.h>
#include <math.h>

// ---------------------------------------------------------------------------
// CDNA5 (gfx1250) wave32 WMMA types
// ---------------------------------------------------------------------------
typedef __attribute__((ext_vector_type(16))) __bf16 v16bf;
typedef __attribute__((ext_vector_type(8)))  float  v8f;

union AFrag { v16bf v; unsigned u[8]; };   // 16x32 bf16 A-matrix fragment
union BFrag { v16bf v; unsigned u[8]; };   // 32x16 bf16 B-matrix fragment

static __device__ __forceinline__ __bf16 f32_to_bf16(float f) {
    unsigned u = __builtin_bit_cast(unsigned, f);
    u += 0x7FFFu + ((u >> 16) & 1u);       // round-to-nearest-even
    unsigned short h = (unsigned short)(u >> 16);
    return __builtin_bit_cast(__bf16, h);
}

static __device__ __forceinline__ v8f wmma_bf16(const AFrag& a, const BFrag& b, v8f c) {
    return __builtin_amdgcn_wmma_f32_16x16x32_bf16(
        false, a.v, false, b.v, (short)0, c, false, false);
}

// ---------------------------------------------------------------------------
// CDNA5 async global->LDS copies (ASYNCcnt path, CDNA5 ISA).
// Probe-confirmed (round-2 diagnostic): the builtins take a pointer to an
// int-vector of the transfer width, global side in addrspace(1), LDS side in
// addrspace(3). Global flat addresses == as1 numerically; the LDS offset is
// the low 32 bits of the flat shared address (ISA §10.2 aperture mapping),
// so we build the qualified pointers through integer casts.
// ---------------------------------------------------------------------------
typedef int v2i __attribute__((ext_vector_type(2)));
typedef int v4i __attribute__((ext_vector_type(4)));
typedef __attribute__((address_space(1))) v2i v2i_as1;
typedef __attribute__((address_space(3))) v2i v2i_as3;
typedef __attribute__((address_space(1))) v4i v4i_as1;
typedef __attribute__((address_space(3))) v4i v4i_as3;

static __device__ __forceinline__ void async_copy_b64(const void* g, const void* l) {
#if __has_builtin(__builtin_amdgcn_global_load_async_to_lds_b64)
    __builtin_amdgcn_global_load_async_to_lds_b64(
        (v2i_as1*)(unsigned long long)g,
        (v2i_as3*)(unsigned)(unsigned long long)l, 0, 0);
#else
    unsigned laddr = (unsigned)(unsigned long long)l;
    asm volatile("global_load_async_to_lds_b64 %0, %1, off"
                 :: "v"(laddr), "v"(g) : "memory");
#endif
}

static __device__ __forceinline__ void async_copy_b128(const void* g, const void* l) {
#if __has_builtin(__builtin_amdgcn_global_load_async_to_lds_b128)
    __builtin_amdgcn_global_load_async_to_lds_b128(
        (v4i_as1*)(unsigned long long)g,
        (v4i_as3*)(unsigned)(unsigned long long)l, 0, 0);
#else
    unsigned laddr = (unsigned)(unsigned long long)l;
    asm volatile("global_load_async_to_lds_b128 %0, %1, off"
                 :: "v"(laddr), "v"(g) : "memory");
#endif
}

static __device__ __forceinline__ void wait_async0() {
#if __has_builtin(__builtin_amdgcn_s_wait_asynccnt)
    __builtin_amdgcn_s_wait_asynccnt(0);
#else
    asm volatile("s_wait_asynccnt 0x0" ::: "memory");
#endif
}

// ---------------------------------------------------------------------------
// Problem constants
// ---------------------------------------------------------------------------
#define BATCH 4
#define TSEQ  2048
#define CDIM  1024
#define NH    16
#define DHEAD 64
#define BT    (BATCH * TSEQ)

// ---------------------------------------------------------------------------
// Kernel 1: fp32 -> bf16 conversion
// ---------------------------------------------------------------------------
__global__ __launch_bounds__(256)
void k_convert_f32_bf16(const float4* __restrict__ src, uint2* __restrict__ dst, int n4) {
    int i = blockIdx.x * 256 + threadIdx.x;
    if (i < n4) {
        float4 f = src[i];
        union { uint2 u; __bf16 b[4]; } t;
        t.b[0] = f32_to_bf16(f.x);
        t.b[1] = f32_to_bf16(f.y);
        t.b[2] = f32_to_bf16(f.z);
        t.b[3] = f32_to_bf16(f.w);
        dst[i] = t.u;
    }
}

// ---------------------------------------------------------------------------
// Kernel 2: double-buffered bf16 WMMA GEMM, async A staging, pipelined B.
// C[M,N] = A[M,K] * B[K,N]; block tile 128x128, K-step 32, 8 waves,
// each wave owns 64x32 (4x2 WMMA accumulators).
// ---------------------------------------------------------------------------
#define BM 128
#define BN 128
#define BK 32
#define LDA 40
#define LDB 40

template <bool OUT_BF16>
__global__ __launch_bounds__(256)
void k_gemm_bf16(const __bf16* __restrict__ A, const __bf16* __restrict__ B,
                 void* __restrict__ Cout, int M, int N, int K) {
    __shared__ __bf16 As[2][BM * LDA];       // As[buf][m][k]
    __shared__ __bf16 Bs[2][BN * LDB];       // Bs[buf][n][k] (transposed tile)

    const int tid  = threadIdx.x;
    const int lane = tid & 31;
    const int wave = tid >> 5;
    const int lm   = lane & 15;
    const int half = lane >> 4;
    const int wm   = (wave & 1) * 64;
    const int wn   = (wave >> 1) * 32;

    const int m0 = blockIdx.y * BM;
    const int n0 = blockIdx.x * BN;

    v8f acc[4][2];
#pragma unroll
    for (int ms = 0; ms < 4; ++ms)
#pragma unroll
        for (int ns = 0; ns < 2; ++ns)
#pragma unroll
            for (int v = 0; v < 8; ++v) acc[ms][ns][v] = 0.0f;

    // per-thread staging coordinates (4 chunks each for A and B tiles)
    int ar[4], asg[4], bkr[4], bnc[4];
#pragma unroll
    for (int t = 0; t < 4; ++t) {
        int s = tid + 256 * t;               // A: 1024 b64 segs (4 bf16 each)
        ar[t]  = s >> 3;
        asg[t] = (s & 7) << 2;
        int c = tid + 256 * t;               // B: 1024 chunks of 4 bf16
        bkr[t] = c >> 5;
        bnc[t] = (c & 31) << 2;
    }

    auto stage_A = [&](int buf, int k0) {
#pragma unroll
        for (int t = 0; t < 4; ++t)
            async_copy_b64(A + (size_t)(m0 + ar[t]) * K + k0 + asg[t],
                           &As[buf][ar[t] * LDA + asg[t]]);
    };
    uint2 br[4];
    auto load_B = [&](int k0) {
#pragma unroll
        for (int t = 0; t < 4; ++t)
            br[t] = *(const uint2*)(B + (size_t)(k0 + bkr[t]) * N + n0 + bnc[t]);
    };
    auto store_B = [&](int buf) {
#pragma unroll
        for (int t = 0; t < 4; ++t) {
            union { uint2 u; __bf16 b[4]; } x; x.u = br[t];
            Bs[buf][(bnc[t] + 0) * LDB + bkr[t]] = x.b[0];
            Bs[buf][(bnc[t] + 1) * LDB + bkr[t]] = x.b[1];
            Bs[buf][(bnc[t] + 2) * LDB + bkr[t]] = x.b[2];
            Bs[buf][(bnc[t] + 3) * LDB + bkr[t]] = x.b[3];
        }
    };

    // prologue: stage tile 0
    stage_A(0, 0);
    load_B(0);
    store_B(0);
    wait_async0();
    __syncthreads();

    int buf = 0;
    for (int k0 = 0; k0 < K; k0 += BK) {
        const bool more = (k0 + BK) < K;
        if (more) { stage_A(buf ^ 1, k0 + BK); load_B(k0 + BK); }

        // ---- A fragments (16-bit A 16x32 layout) ------------------------
        const __bf16* asb = As[buf];
        const __bf16* bsb = Bs[buf];
        AFrag af[4];
#pragma unroll
        for (int ms = 0; ms < 4; ++ms) {
            const __bf16* arow = asb + (wm + ms * 16 + lm) * LDA;
#pragma unroll
            for (int v = 0; v < 8; ++v) {
                int kk = (half << 4) + (v << 1);
                af[ms].u[v] = *(const unsigned*)(arow + kk);
            }
        }
        // ---- B fragments (16-bit B 32x16 layout) ------------------------
        BFrag bf[2];
#pragma unroll
        for (int ns = 0; ns < 2; ++ns) {
            const __bf16* brow = bsb + (wn + ns * 16 + lm) * LDB;
#pragma unroll
            for (int v = 0; v < 8; ++v)
                bf[ns].u[v] = *(const unsigned*)(brow + (half << 4) + (v << 1));
        }
#pragma unroll
        for (int ms = 0; ms < 4; ++ms)
#pragma unroll
            for (int ns = 0; ns < 2; ++ns)
                acc[ms][ns] = wmma_bf16(af[ms], bf[ns], acc[ms][ns]);

        if (more) store_B(buf ^ 1);
        wait_async0();
        __syncthreads();
        buf ^= 1;
    }

    // ---- epilogue: C layout  m = v + 8*half, n = lm ---------------------
#pragma unroll
    for (int ms = 0; ms < 4; ++ms) {
#pragma unroll
        for (int ns = 0; ns < 2; ++ns) {
#pragma unroll
            for (int v = 0; v < 8; ++v) {
                int row = m0 + wm + ms * 16 + v + (half << 3);
                int col = n0 + wn + ns * 16 + lm;
                float val = acc[ms][ns][v];
                if constexpr (OUT_BF16)
                    ((__bf16*)Cout)[(size_t)row * N + col] = f32_to_bf16(val);
                else
                    ((float*)Cout)[(size_t)row * N + col] = val;
            }
        }
    }
}

// ---------------------------------------------------------------------------
// Kernel 3: causal flash attention, bf16 WMMA, f32 online softmax.
// grid = (TSEQ/64, BATCH*NH), block = 128 (4 waves, 16 q-rows per wave).
// Double-buffered K/V tiles: K staged via async B128 DMA, V pipelined
// through registers with an LDS transpose.
// ---------------------------------------------------------------------------
__global__ __launch_bounds__(128)
void k_flash_attn(const __bf16* __restrict__ qkv, __bf16* __restrict__ y) {
    const int qb   = blockIdx.x;
    const int b    = blockIdx.y / NH;
    const int h    = blockIdx.y % NH;
    const int tid  = threadIdx.x;
    const int lane = tid & 31;
    const int w    = tid >> 5;
    const int lm   = lane & 15;
    const int half = lane >> 4;

    __shared__ __bf16 Ks[2][64 * 72];        // Ks[buf][key][d]
    __shared__ __bf16 Vt[2][64 * 72];        // Vt[buf][d][key]
    __shared__ __bf16 Ps[4][16 * 72];        // per-wave P tile

    const int q0w = qb * 64 + w * 16;
    const size_t rstride = 3 * (size_t)CDIM;
    const __bf16* base = qkv + (size_t)(b * TSEQ) * rstride + h * DHEAD;

    // ---- Q as two A-fragments straight from global ----------------------
    AFrag qf[2];
#pragma unroll
    for (int ks = 0; ks < 2; ++ks) {
        const __bf16* qrow = base + (size_t)(q0w + lm) * rstride + ks * 32;
#pragma unroll
        for (int v = 0; v < 8; ++v) {
            int kk = (half << 4) + (v << 1);
            qf[ks].u[v] = *(const unsigned*)(qrow + kk);
        }
    }

    auto stage_K = [&](int buf, int j) {     // async DMA, 512 x 16B segments
        const __bf16* kb = base + (size_t)(j * 64) * rstride + CDIM;
#pragma unroll
        for (int t = 0; t < 4; ++t) {
            int c  = tid + t * 128;
            int kr = c >> 3;
            int sg = (c & 7) << 3;           // bf16 offset, 16B aligned
            async_copy_b128(kb + (size_t)kr * rstride + sg,
                            &Ks[buf][kr * 72 + sg]);
        }
    };
    unsigned vr[16];
    auto load_V = [&](int j) {
        const __bf16* vb = base + (size_t)(j * 64) * rstride + 2 * CDIM;
#pragma unroll
        for (int t = 0; t < 16; ++t) {
            int c = tid + t * 128;
            vr[t] = *(const unsigned*)(vb + (size_t)(c >> 5) * rstride + (c & 31) * 2);
        }
    };
    auto store_V = [&](int buf) {
#pragma unroll
        for (int t = 0; t < 16; ++t) {
            int c = tid + t * 128;
            int kr = c >> 5, dc2 = (c & 31) * 2;
            union { unsigned u; __bf16 bb[2]; } x; x.u = vr[t];
            Vt[buf][(dc2 + 0) * 72 + kr] = x.bb[0];
            Vt[buf][(dc2 + 1) * 72 + kr] = x.bb[1];
        }
    };

    float mrun[8], lrun[8];
    v8f   Oacc[4];
#pragma unroll
    for (int v = 0; v < 8; ++v) { mrun[v] = -3.0e38f; lrun[v] = 0.0f; }
#pragma unroll
    for (int d4 = 0; d4 < 4; ++d4)
#pragma unroll
        for (int v = 0; v < 8; ++v) Oacc[d4][v] = 0.0f;

    // prologue: stage block 0
    stage_K(0, 0);
    load_V(0);
    store_V(0);
    wait_async0();
    __syncthreads();

    int buf = 0;
    for (int j = 0; j <= qb; ++j) {
        const int  kbase = j * 64;
        const bool more  = j < qb;
        if (more) { stage_K(buf ^ 1, j + 1); load_V(j + 1); }

        // ---- S = Q*K^T ---------------------------------------------------
        const __bf16* ksb = Ks[buf];
        const __bf16* vtb = Vt[buf];
        v8f S[4];
#pragma unroll
        for (int ns = 0; ns < 4; ++ns) {
#pragma unroll
            for (int v = 0; v < 8; ++v) S[ns][v] = 0.0f;
#pragma unroll
            for (int ks = 0; ks < 2; ++ks) {
                BFrag bk;
                const __bf16* krow = ksb + (ns * 16 + lm) * 72 + ks * 32;
#pragma unroll
                for (int v = 0; v < 8; ++v)
                    bk.u[v] = *(const unsigned*)(krow + (half << 4) + (v << 1));
                S[ns] = wmma_bf16(qf[ks], bk, S[ns]);
            }
        }

        // ---- online softmax (C layout: row = v + 8*half, col = lm) ------
#pragma unroll
        for (int v = 0; v < 8; ++v) {
            const int qg = q0w + v + (half << 3);
            float sv[4];
            float rmax = -3.0e38f;
#pragma unroll
            for (int ns = 0; ns < 4; ++ns) {
                float x = S[ns][v] * 0.125f;
                int kg = kbase + ns * 16 + lm;
                if (kg > qg) x = -3.0e38f;
                sv[ns] = x;
                rmax = fmaxf(rmax, x);
            }
#pragma unroll
            for (int off = 8; off >= 1; off >>= 1)
                rmax = fmaxf(rmax, __shfl_xor(rmax, off, 32));
            const float mnew = fmaxf(mrun[v], rmax);
            const float corr = __expf(mrun[v] - mnew);
            float rsum = 0.0f;
            __bf16* prow = &Ps[w][(v + (half << 3)) * 72];
#pragma unroll
            for (int ns = 0; ns < 4; ++ns) {
                float p = __expf(sv[ns] - mnew);
                rsum += p;
                prow[ns * 16 + lm] = f32_to_bf16(p);
            }
#pragma unroll
            for (int off = 8; off >= 1; off >>= 1)
                rsum += __shfl_xor(rsum, off, 32);
            lrun[v] = lrun[v] * corr + rsum;
            mrun[v] = mnew;
#pragma unroll
            for (int d4 = 0; d4 < 4; ++d4) Oacc[d4][v] *= corr;
        }

        // ---- O += P*V ----------------------------------------------------
#pragma unroll
        for (int ks = 0; ks < 2; ++ks) {
            AFrag ap;
            const __bf16* prow = &Ps[w][lm * 72 + ks * 32];
#pragma unroll
            for (int v = 0; v < 8; ++v) {
                int kk = (half << 4) + (v << 1);
                ap.u[v] = *(const unsigned*)(prow + kk);
            }
#pragma unroll
            for (int d4 = 0; d4 < 4; ++d4) {
                BFrag bv;
                const __bf16* vrow = vtb + (d4 * 16 + lm) * 72 + ks * 32;
#pragma unroll
                for (int v = 0; v < 8; ++v)
                    bv.u[v] = *(const unsigned*)(vrow + (half << 4) + (v << 1));
                Oacc[d4] = wmma_bf16(ap, bv, Oacc[d4]);
            }
        }

        if (more) store_V(buf ^ 1);
        wait_async0();
        __syncthreads();
        buf ^= 1;
    }

    // ---- write y --------------------------------------------------------
#pragma unroll
    for (int d4 = 0; d4 < 4; ++d4) {
#pragma unroll
        for (int v = 0; v < 8; ++v) {
            const int qg = q0w + v + (half << 3);
            float o = Oacc[d4][v] / lrun[v];
            y[(size_t)(b * TSEQ + qg) * CDIM + h * DHEAD + d4 * 16 + lm] = f32_to_bf16(o);
        }
    }
}

// ---------------------------------------------------------------------------
// Host-side launcher
// ---------------------------------------------------------------------------
extern "C" void kernel_launch(void* const* d_in, const int* in_sizes, int n_in,
                              void* d_out, int out_size, void* d_ws, size_t ws_size,
                              hipStream_t stream) {
    const float* x     = (const float*)d_in[0];   // [4,2048,1024]
    const float* wqkv  = (const float*)d_in[1];   // [1024,3072]
    const float* wproj = (const float*)d_in[2];   // [1024,1024]
    float* out = (float*)d_out;                   // [4,2048,1024] fp32

    char* ws = (char*)d_ws;
    size_t off = 0;
    auto alloc = [&](size_t bytes) { char* p = ws + off; off = (off + bytes + 255) & ~(size_t)255; return p; };
    __bf16* xb   = (__bf16*)alloc((size_t)BT * CDIM * 2);
    __bf16* wqb  = (__bf16*)alloc((size_t)CDIM * 3 * CDIM * 2);
    __bf16* wpb  = (__bf16*)alloc((size_t)CDIM * CDIM * 2);
    __bf16* qkvb = (__bf16*)alloc((size_t)BT * 3 * CDIM * 2);
    __bf16* yb   = (__bf16*)alloc((size_t)BT * CDIM * 2);

    {
        int n4 = (BT * CDIM) / 4;
        k_convert_f32_bf16<<<(n4 + 255) / 256, 256, 0, stream>>>((const float4*)x, (uint2*)xb, n4);
        n4 = (CDIM * 3 * CDIM) / 4;
        k_convert_f32_bf16<<<(n4 + 255) / 256, 256, 0, stream>>>((const float4*)wqkv, (uint2*)wqb, n4);
        n4 = (CDIM * CDIM) / 4;
        k_convert_f32_bf16<<<(n4 + 255) / 256, 256, 0, stream>>>((const float4*)wproj, (uint2*)wpb, n4);
    }

    k_gemm_bf16<true><<<dim3((3 * CDIM) / BN, BT / BM), 256, 0, stream>>>(
        xb, wqb, (void*)qkvb, BT, 3 * CDIM, CDIM);

    k_flash_attn<<<dim3(TSEQ / 64, BATCH * NH), 128, 0, stream>>>(qkvb, yb);

    k_gemm_bf16<false><<<dim3(CDIM / BN, BT / BM), 256, 0, stream>>>(
        yb, wpb, (void*)out, BT, CDIM, CDIM);
}
